// TransducerJoint_29128468202006
// MI455X (gfx1250) — matrix operand order repaired and checked by reading, not compiled, not run on hardware
//
#include <hip/hip_runtime.h>
#include <hip/hip_bf16.h>
#include <math.h>

typedef _Float16 half_t;
typedef __attribute__((ext_vector_type(16))) _Float16 v16h;
typedef __attribute__((ext_vector_type(8)))  _Float16 v8h;
typedef __attribute__((ext_vector_type(8)))  float    v8f;
typedef __attribute__((ext_vector_type(4)))  float    v4f;
typedef __attribute__((ext_vector_type(4)))  _Float16 v4h;

#define TJ_B 8
#define TJ_T 200
#define TJ_U 100
#define TJ_D 640
#define TJ_J 640
#define TJ_V 1024
#define TJ_ROWS (TJ_B * TJ_T * TJ_U)     // 160000
#define TJ_CHUNK 32000                   // rows per chunk (5 chunks); 41MB f16 act fits L2

#if __has_builtin(__builtin_amdgcn_sched_group_barrier)
#define TJ_SG(mask, size, id) __builtin_amdgcn_sched_group_barrier(mask, size, id)
#else
#define TJ_SG(mask, size, id)
#endif
// masks: 0x020 = VMEM read, 0x008 = MFMA/WMMA

__device__ __forceinline__ float tj_tanh(float x) {
#if __has_builtin(__builtin_amdgcn_tanhf)
  return __builtin_amdgcn_tanhf(x);
#else
  return tanhf(x);
#endif
}

union tj_frag { v16h v; v8h h[2]; };

// A fragment (16x32 f16) at (m0,k0) from row-major [*,lda]:
// lanes 0-15 (M=lane): halves k0+0..7 and k0+16..23 ; lanes 16-31: +8
__device__ __forceinline__ v16h tj_loadA(const half_t* __restrict__ A, int m0, int k0,
                                         int lda, int laneM, int laneHi) {
  const half_t* ap = A + (size_t)(m0 + laneM) * lda + k0 + laneHi * 8;
  tj_frag u;
  u.h[0] = *(const v8h*)ap;
  u.h[1] = *(const v8h*)(ap + 16);
  return u.v;
}

// B fragment (32x16 f16) at (k0,n0) from transposed row-major BT[N,ldk]:
// lanes 0-15 (N=lane): halves k0+0..7 and k0+16..23 ; lanes 16-31: +8
__device__ __forceinline__ v16h tj_loadB(const half_t* __restrict__ BT, int n0, int k0,
                                         int ldk, int laneM, int laneHi) {
  const half_t* bp = BT + (size_t)(n0 + laneM) * ldk + k0 + laneHi * 8;
  tj_frag u;
  u.h[0] = *(const v8h*)bp;
  u.h[1] = *(const v8h*)(bp + 16);
  return u.v;
}

// ---------------- elementwise f32 -> f16 convert (4 elems/thread) ----------------
__global__ void tj_cvt_f16(const float* __restrict__ src, half_t* __restrict__ dst, unsigned n) {
  unsigned i = (blockIdx.x * blockDim.x + threadIdx.x) * 4u;
  if (i >= n) return;
  v4f s = *(const v4f*)(src + i);
  v4h d;
  d[0] = (half_t)s[0]; d[1] = (half_t)s[1]; d[2] = (half_t)s[2]; d[3] = (half_t)s[3];
  *(v4h*)(dst + i) = d;
}

// ---------------- transpose + convert: dst[c*rows + r] = src[r*cols + c] ----------------
__global__ void tj_transpose_f16(const float* __restrict__ src, half_t* __restrict__ dst,
                                 int rows, int cols) {
  unsigned idx = blockIdx.x * blockDim.x + threadIdx.x;
  unsigned total = (unsigned)rows * (unsigned)cols;
  if (idx >= total) return;
  unsigned r = idx / (unsigned)cols;
  unsigned c = idx % (unsigned)cols;
  dst[(size_t)c * rows + r] = (half_t)src[idx];
}

// ---------------- prejoint GEMM: C_h[M,N] = f16( A_h[M,K] @ B (given as BT_h[N,K]) + bias ) ----
// block: 256 thr = 8 waves; wave w -> 16 rows (mBase + 16w) x 64 cols (4 n-tiles)
__global__ __launch_bounds__(256) void tj_prejoint_gemm(
    const half_t* __restrict__ A, const half_t* __restrict__ BT,
    const float* __restrict__ bias, half_t* __restrict__ C,
    int M, int K, int N) {
  const int wave = threadIdx.x >> 5;
  const int lane = threadIdx.x & 31;
  const int laneM = lane & 15;
  const int laneHi = lane >> 4;
  const int mBase = blockIdx.x * 128 + wave * 16;
  const int nBase = blockIdx.y * 64;

  int arow = mBase + laneM;
  if (arow >= M) arow = M - 1;            // clamp (data only; EXEC stays full)

  v8f acc[4] = {v8f{}, v8f{}, v8f{}, v8f{}};

#pragma unroll 2
  for (int k0 = 0; k0 < K; k0 += 32) {
    v16h a = tj_loadA(A, arow, k0, K, 0, laneHi);
    v16h b[4];
#pragma unroll
    for (int j = 0; j < 4; ++j)
      b[j] = tj_loadB(BT, nBase + j * 16, k0, K, laneM, laneHi);
#pragma unroll
    for (int j = 0; j < 4; ++j)
      acc[j] = __builtin_amdgcn_wmma_f32_16x16x32_f16(
          false, a, false, b[j], (short)0, acc[j], false, false);
    TJ_SG(0x020, 10, 0);   // all loads of this K-step first
    TJ_SG(0x008, 4, 0);    // then the 4 WMMAs
  }

#pragma unroll
  for (int j = 0; j < 4; ++j) {
    const int n = nBase + j * 16 + laneM;
    const float bv = bias ? bias[n] : 0.0f;
#pragma unroll
    for (int r = 0; r < 8; ++r) {
      const int m = mBase + laneHi * 8 + r;
      if (m < M) C[(size_t)m * N + n] = (half_t)(acc[j][r] + bv);
    }
  }
}

// ---------------- act pass: act_h[r,k] = f16(tanh(ENC[encRow,k] + PRED[predRow,k])) ----------
__global__ void tj_act(const half_t* __restrict__ ENC, const half_t* __restrict__ PRED,
                       half_t* __restrict__ act, int rowStart, int nRows) {
  unsigned idx = (blockIdx.x * blockDim.x + threadIdx.x) * 8u;
  unsigned total = (unsigned)nRows * TJ_J;
  if (idx >= total) return;
  unsigned r = idx / TJ_J;
  unsigned k = idx % TJ_J;
  unsigned grow = (unsigned)rowStart + r;
  unsigned b   = grow / (TJ_T * TJ_U);
  unsigned rem = grow % (TJ_T * TJ_U);
  unsigned t = rem / TJ_U;
  unsigned u = rem % TJ_U;
  const half_t* ep = ENC  + ((size_t)(b * TJ_T + t)) * TJ_J + k;
  const half_t* pp = PRED + ((size_t)(b * TJ_U + u)) * TJ_J + k;
  v8h e = *(const v8h*)ep;
  v8h p = *(const v8h*)pp;
  v8h o;
#pragma unroll
  for (int i = 0; i < 8; ++i) {
    float x = (float)e[i] + (float)p[i];
    o[i] = (half_t)tj_tanh(x);
  }
  *(v8h*)(act + idx) = o;
}

// ---------------- joint GEMM: out[M,1024] = f32( act[M,640] @ W_out (as WoutT[1024,640]) + b_out )
// block tile 256x128; 8 waves as 4 m-groups x 2 n-groups; wave = 4 m-tiles x 4 n-tiles (16 WMMA)
// Software-pipelined: B fragments double-buffered one K-iteration ahead.
__global__ __launch_bounds__(256) void tj_joint_gemm(
    const half_t* __restrict__ A,      // [Mchunk, 640]
    const half_t* __restrict__ BT,     // [1024, 640]
    const float* __restrict__ bias,    // [1024]
    float* __restrict__ C,             // [Mchunk, 1024]
    int Mchunk) {
  const int K = TJ_J;
  const int NSTEP = K / 32;                  // 20
  const int wave = threadIdx.x >> 5;
  const int lane = threadIdx.x & 31;
  const int laneM = lane & 15;
  const int laneHi = lane >> 4;
  const int mg = wave >> 1;                  // 0..3
  const int ng = wave & 1;                   // 0..1
  const int mBase = blockIdx.y * 256 + mg * 64;
  const int nBase = blockIdx.x * 128 + ng * 64;
  (void)Mchunk;

  v8f acc[4][4];
#pragma unroll
  for (int i = 0; i < 4; ++i)
#pragma unroll
    for (int j = 0; j < 4; ++j) acc[i][j] = v8f{};

  // prologue: B fragments for K-step 0
  v16h bcur[4];
#pragma unroll
  for (int j = 0; j < 4; ++j)
    bcur[j] = tj_loadB(BT, nBase + j * 16, 0, K, laneM, laneHi);

#pragma unroll 2
  for (int ks = 0; ks < NSTEP; ++ks) {
    const int k0 = ks * 32;
    const int kn = (ks == NSTEP - 1) ? 0 : k0 + 32;  // last prefetch: harmless re-read of k=0
    v16h a[4], bnxt[4];
#pragma unroll
    for (int i = 0; i < 4; ++i)
      a[i] = tj_loadA(A, mBase + i * 16, k0, K, laneM, laneHi);
#pragma unroll
    for (int j = 0; j < 4; ++j)
      bnxt[j] = tj_loadB(BT, nBase + j * 16, kn, K, laneM, laneHi);

#pragma unroll
    for (int j = 0; j < 4; ++j)
#pragma unroll
      for (int i = 0; i < 4; ++i)
        acc[i][j] = __builtin_amdgcn_wmma_f32_16x16x32_f16(
            false, a[i], false, bcur[j], (short)0, acc[i][j], false, false);

#pragma unroll
    for (int j = 0; j < 4; ++j) bcur[j] = bnxt[j];

    // issue pattern per K-step: 8 A-loads, then WMMA quads with next-iter B-loads interleaved
    TJ_SG(0x020, 8, 0);   // a0..a3 (this step)
    TJ_SG(0x008, 4, 0);
    TJ_SG(0x020, 2, 0);   // bnxt0
    TJ_SG(0x008, 4, 0);
    TJ_SG(0x020, 2, 0);   // bnxt1
    TJ_SG(0x008, 4, 0);
    TJ_SG(0x020, 2, 0);   // bnxt2
    TJ_SG(0x008, 4, 0);
    TJ_SG(0x020, 2, 0);   // bnxt3
  }

#pragma unroll
  for (int i = 0; i < 4; ++i) {
#pragma unroll
    for (int j = 0; j < 4; ++j) {
      const int n = nBase + j * 16 + laneM;
      const float bv = bias[n];
#pragma unroll
      for (int r = 0; r < 8; ++r) {
        const int m = mBase + i * 16 + laneHi * 8 + r;
        // streaming output: non-temporal keeps act/W resident in L2
        __builtin_nontemporal_store(acc[i][j][r] + bv, &C[(size_t)m * TJ_V + n]);
      }
    }
  }
}

extern "C" void kernel_launch(void* const* d_in, const int* in_sizes, int n_in,
                              void* d_out, int out_size, void* d_ws, size_t ws_size,
                              hipStream_t stream) {
  (void)in_sizes; (void)n_in; (void)out_size; (void)ws_size;
  const float* enc_out = (const float*)d_in[0];   // [8,200,640]
  const float* pred_out = (const float*)d_in[1];  // [8,100,640]
  const float* W_enc  = (const float*)d_in[2];    // [640,640]
  const float* b_enc  = (const float*)d_in[3];    // [640]
  const float* W_pred = (const float*)d_in[4];    // [640,640]
  const float* W_out  = (const float*)d_in[5];    // [640,1024]
  const float* b_out  = (const float*)d_in[6];    // [1024]
  float* out = (float*)d_out;                     // [8,200,100,1024]

  // workspace carve-up (element counts; all 16B aligned)
  half_t* ws = (half_t*)d_ws;
  const size_t N_ENCIN  = (size_t)TJ_B * TJ_T * TJ_D;   // 1,024,000
  const size_t N_PREDIN = (size_t)TJ_B * TJ_U * TJ_D;   //   512,000
  const size_t N_WENC   = (size_t)TJ_D * TJ_J;          //   409,600
  const size_t N_WOUT   = (size_t)TJ_J * TJ_V;          //   655,360
  half_t* enc_h   = ws;
  half_t* pred_h  = enc_h  + N_ENCIN;
  half_t* WencT   = pred_h + N_PREDIN;
  half_t* WpredT  = WencT  + N_WENC;
  half_t* WoutT   = WpredT + N_WENC;
  half_t* ENC     = WoutT  + N_WOUT;                    // [1600, 640] f16
  half_t* PRED    = ENC    + N_ENCIN;                   // [800, 640] f16
  half_t* act     = PRED   + N_PREDIN;                  // [32000, 640] f16 chunk

  // 1) convert activations to f16
  tj_cvt_f16<<<(unsigned)(N_ENCIN / 4 + 255) / 256, 256, 0, stream>>>(enc_out, enc_h, (unsigned)N_ENCIN);
  tj_cvt_f16<<<(unsigned)(N_PREDIN / 4 + 255) / 256, 256, 0, stream>>>(pred_out, pred_h, (unsigned)N_PREDIN);

  // 2) transpose weights to [N,K] f16 so B-fragments are contiguous per lane
  tj_transpose_f16<<<(unsigned)(N_WENC + 255) / 256, 256, 0, stream>>>(W_enc, WencT, TJ_D, TJ_J);
  tj_transpose_f16<<<(unsigned)(N_WENC + 255) / 256, 256, 0, stream>>>(W_pred, WpredT, TJ_D, TJ_J);
  tj_transpose_f16<<<(unsigned)(N_WOUT + 255) / 256, 256, 0, stream>>>(W_out, WoutT, TJ_J, TJ_V);

  // 3) prejoint GEMMs (WMMA f16, f32 accum), results stored f16
  {
    dim3 grid((1600 + 127) / 128, TJ_J / 64);
    tj_prejoint_gemm<<<grid, 256, 0, stream>>>(enc_h, WencT, b_enc, ENC, 1600, TJ_D, TJ_J);
  }
  {
    dim3 grid((800 + 127) / 128, TJ_J / 64);
    tj_prejoint_gemm<<<grid, 256, 0, stream>>>(pred_h, WpredT, nullptr, PRED, 800, TJ_D, TJ_J);
  }

  // 4) chunked: act materialization (fits L2) + big WMMA GEMM
  const int nChunks = TJ_ROWS / TJ_CHUNK;               // 5
  for (int c = 0; c < nChunks; ++c) {
    const int rowStart = c * TJ_CHUNK;
    const unsigned actThreads = (unsigned)TJ_CHUNK * TJ_J / 8;   // 2.56M
    tj_act<<<actThreads / 256, 256, 0, stream>>>(ENC, PRED, act, rowStart, TJ_CHUNK);
    dim3 grid(TJ_V / 128, TJ_CHUNK / 256);              // (8, 125)
    tj_joint_gemm<<<grid, 256, 0, stream>>>(act, WoutT, b_out,
                                            out + (size_t)rowStart * TJ_V, TJ_CHUNK);
  }
}
